// SessionGraphGNN_17394617549172
// MI455X (gfx1250) — matrix operand order repaired and checked by reading, not compiled, not run on hardware
//
#include <hip/hip_runtime.h>
#include <hip/hip_bf16.h>
#include <math.h>

// ---------------------------------------------------------------------------
// Problem constants (match reference)
// ---------------------------------------------------------------------------
#define NN   20000
#define EE   320000
#define EP   (EE + NN)          // edges incl. self loops
#define GG   64
#define F_IN 32
#define HID  64
#define H1   4
#define HC1  (H1 * HID)         // 256

typedef __attribute__((ext_vector_type(16))) _Float16 v16h;
typedef __attribute__((ext_vector_type(8)))  float    v8f;

// ---------------------------------------------------------------------------
// WMMA GEMM:  C[M,Nc] = A[M,K] @ B[K,Nc]   (f32 in/out, f16 MACs)
// Block = 4 waves; block stages a 64-column B panel into LDS (f16, column-
// major per 16-col tile) with coalesced global reads, then each wave computes
// one 16x16 tile, reading B fragments as 32B ds_load_b128 pairs.
//
// Fragment layouts used by this kernel (wave32):
//  A (16x32 f16): lane l -> row (l&15); hs=l>>4:  j -> K = k0+hs*16+j  (16 contiguous)
//  B (32x16 f16): lane l -> col (l&15); j -> K = k0+hs*16+j  (16 contiguous)
//  C/D (16x16 f32): VGPR i -> row hs*8+i, col (l&15)
// ---------------------------------------------------------------------------
#define WPB 4
__global__ void gemm_wmma_f16(const float* __restrict__ A,
                              const float* __restrict__ B,
                              float* __restrict__ C,
                              int M, int K, int Nc) {
  extern __shared__ _Float16 bshT[];      // [WPB*16 cols][K]  (col-major per tile)
  const int lane   = threadIdx.x & 31;
  const int wave   = threadIdx.x >> 5;
  const int mtile  = blockIdx.x;
  const int ncol0  = blockIdx.y * (WPB * 16);   // first column of block's panel

  // ---- cooperative stage: B[k, ncol0 .. ncol0+63] -> LDS f16 (coalesced) ----
  const int panel = WPB * 16 * K;
  for (int idx = threadIdx.x; idx < panel; idx += blockDim.x) {
    int cg = idx & (WPB * 16 - 1);        // 0..63
    int k  = idx >> 6;                    // WPB*16 == 64
    int gc = ncol0 + cg;
    int gcc = gc < Nc ? gc : Nc - 1;      // clamp (values unused when OOB)
    bshT[(long long)cg * K + k] = (_Float16)B[(long long)k * Nc + gcc];
  }
  __syncthreads();

  const int ntile = blockIdx.y * WPB + wave;
  if (ntile * 16 >= Nc || mtile * 16 >= M) return;   // wave-uniform exit

  const int r  = lane & 15;
  const int hs = lane >> 4;
  const int row0 = mtile * 16;

  const _Float16* bbase = bshT + (long long)(wave * 16 + r) * K;

  v8f acc = {};
  for (int k0 = 0; k0 < K; k0 += 32) {
    v16h a;
    const float* arow = A + (long long)(row0 + r) * K + k0 + hs * 16;
#pragma unroll
    for (int j = 0; j < 8; ++j)  a[j]     = (_Float16)arow[j];
#pragma unroll
    for (int j = 0; j < 8; ++j)  a[8 + j] = (_Float16)arow[8 + j];
    v16h b = *(const v16h*)(bbase + k0 + hs * 16);   // 32B aligned: 2x ds_load_b128
    acc = __builtin_amdgcn_wmma_f32_16x16x32_f16(false, a, false, b,
                                                 (short)0, acc, false, false);
  }
  float* crow = C + (long long)(row0 + hs * 8) * Nc + ntile * 16 + r;
#pragma unroll
  for (int i = 0; i < 8; ++i) crow[(long long)i * Nc] = acc[i];
}

// ---------------------------------------------------------------------------
// Utility kernels
// ---------------------------------------------------------------------------
__global__ void fill_f32(float* __restrict__ p, float v, long long n) {
  long long i = blockIdx.x * (long long)blockDim.x + threadIdx.x;
  if (i < n) p[i] = v;
}

__device__ __forceinline__ void atomicMaxF(float* addr, float val) {
  if (val > 0.0f) atomicMax((int*)addr, __float_as_int(val));
  else            atomicMin((unsigned int*)addr, (unsigned int)__float_as_int(val));
}

__device__ __forceinline__ void edge_sd(const int* __restrict__ srcA,
                                        const int* __restrict__ dstA,
                                        long long eid, int& s, int& d) {
  if (eid < EE) { s = srcA[eid]; d = dstA[eid]; }
  else          { s = d = (int)(eid - EE); }
}

// attention logits:  alS[n,H] = <h[n,h,:], a_src[h,:]>, same for alD
template <int H, int C>
__global__ void attn_logits(const float* __restrict__ hlin,
                            const float* __restrict__ a_src,
                            const float* __restrict__ a_dst,
                            float* __restrict__ alS, float* __restrict__ alD) {
  int idx = blockIdx.x * blockDim.x + threadIdx.x;
  if (idx >= NN * H) return;
  int node = idx / H, h = idx % H;
  const float* hp = hlin + (long long)node * H * C + (long long)h * C;
  const float* as = a_src + h * C;
  const float* ad = a_dst + h * C;
  float s = 0.f, d = 0.f;
#pragma unroll 4
  for (int c = 0; c < C; ++c) { s += hp[c] * as[c]; d += hp[c] * ad[c]; }
  alS[idx] = s; alD[idx] = d;
}

template <int H>
__global__ void edge_max(const int* __restrict__ srcA, const int* __restrict__ dstA,
                         const float* __restrict__ alS, const float* __restrict__ alD,
                         float* __restrict__ m) {
  long long idx = blockIdx.x * (long long)blockDim.x + threadIdx.x;
  if (idx >= (long long)EP * H) return;
  long long eid = idx / H; int h = (int)(idx % H);
  int s, d; edge_sd(srcA, dstA, eid, s, d);
  float e = alS[(long long)s * H + h] + alD[(long long)d * H + h];
  e = e > 0.f ? e : 0.2f * e;                       // leaky_relu(0.2)
  atomicMaxF(&m[(long long)d * H + h], e);
}

template <int H>
__global__ void edge_expsum(const int* __restrict__ srcA, const int* __restrict__ dstA,
                            const float* __restrict__ alS, const float* __restrict__ alD,
                            const float* __restrict__ m, float* __restrict__ ex,
                            float* __restrict__ den) {
  long long idx = blockIdx.x * (long long)blockDim.x + threadIdx.x;
  if (idx >= (long long)EP * H) return;
  long long eid = idx / H; int h = (int)(idx % H);
  int s, d; edge_sd(srcA, dstA, eid, s, d);
  float e = alS[(long long)s * H + h] + alD[(long long)d * H + h];
  e = e > 0.f ? e : 0.2f * e;
  float v = __expf(e - m[(long long)d * H + h]);
  ex[idx] = v;
  atomicAdd(&den[(long long)d * H + h], v);
}

// in-place: ex[e,h] -> alpha[e,h] = ex / (den[dst,h] + eps)
template <int H>
__global__ void alpha_div(const int* __restrict__ srcA, const int* __restrict__ dstA,
                          float* __restrict__ ex, const float* __restrict__ den) {
  long long idx = blockIdx.x * (long long)blockDim.x + threadIdx.x;
  if (idx >= (long long)EP * H) return;
  long long eid = idx / H; int h = (int)(idx % H);
  int s, d; edge_sd(srcA, dstA, eid, s, d);
  ex[idx] = ex[idx] / (den[(long long)d * H + h] + 1e-16f);
}

// one wave per edge: agg[dst, :] += h[src, :] * alpha[edge, head(c)]
template <int H, int C>
__global__ void edge_aggregate(const int* __restrict__ srcA, const int* __restrict__ dstA,
                               const float* __restrict__ hlin,
                               const float* __restrict__ alpha,
                               float* __restrict__ agg) {
  constexpr int HC = H * C;
  const int lane = threadIdx.x & 31;
  long long edge = blockIdx.x * (long long)(blockDim.x >> 5) + (threadIdx.x >> 5);
  if (edge >= EP) return;                          // wave-uniform
  int s, d; edge_sd(srcA, dstA, edge, s, d);
  const float* hs_ = hlin + (long long)s * HC;
  float* ad = agg + (long long)d * HC;
#pragma unroll
  for (int c = lane; c < HC; c += 32) {
    float al = alpha[edge * H + c / C];            // C is constexpr -> shift
    atomicAdd(&ad[c], hs_[c] * al);
  }
}

template <int C>
__global__ void bias_relu(float* __restrict__ x, const float* __restrict__ b) {
  long long idx = blockIdx.x * (long long)blockDim.x + threadIdx.x;
  if (idx >= (long long)NN * C) return;
  float v = x[idx] + b[idx & (C - 1)];
  x[idx] = v > 0.f ? v : 0.f;
}

__global__ void degree_count(const int* __restrict__ srcA, const int* __restrict__ dstA,
                             float* __restrict__ deg) {
  long long eid = blockIdx.x * (long long)blockDim.x + threadIdx.x;
  if (eid >= EP) return;
  int s, d; edge_sd(srcA, dstA, eid, s, d);
  atomicAdd(&deg[d], 1.0f);
}

__global__ void deg_rsqrt(const float* __restrict__ deg, float* __restrict__ dinv) {
  int i = blockIdx.x * blockDim.x + threadIdx.x;
  if (i >= NN) return;
  dinv[i] = rsqrtf(fmaxf(deg[i], 1.0f));
}

// one wave per edge: agg3[dst,:] += h3lin[src,:] * dinv[src]*dinv[dst]
template <int C>
__global__ void gcn_aggregate(const int* __restrict__ srcA, const int* __restrict__ dstA,
                              const float* __restrict__ hlin, const float* __restrict__ dinv,
                              float* __restrict__ agg) {
  const int lane = threadIdx.x & 31;
  long long edge = blockIdx.x * (long long)(blockDim.x >> 5) + (threadIdx.x >> 5);
  if (edge >= EP) return;
  int s, d; edge_sd(srcA, dstA, edge, s, d);
  float norm = dinv[s] * dinv[d];
  const float* hs_ = hlin + (long long)s * C;
  float* ad = agg + (long long)d * C;
#pragma unroll
  for (int c = lane; c < C; c += 32) atomicAdd(&ad[c], hs_[c] * norm);
}

template <int C>
__global__ void pool_sum(const float* __restrict__ h3, const int* __restrict__ batch,
                         float* __restrict__ sums, float* __restrict__ cnts) {
  long long idx = blockIdx.x * (long long)blockDim.x + threadIdx.x;
  if (idx >= (long long)NN * C) return;
  int node = (int)(idx >> 5), c = (int)(idx & (C - 1));   // C == 32
  int g = batch[node];
  atomicAdd(&sums[g * C + c], h3[idx]);
  if (c == 0) atomicAdd(&cnts[g], 1.0f);
}

__global__ void pool_div(const float* __restrict__ sums, const float* __restrict__ cnts,
                         float* __restrict__ pooled) {
  int idx = blockIdx.x * blockDim.x + threadIdx.x;
  if (idx >= GG * 32) return;
  pooled[idx] = sums[idx] / fmaxf(cnts[idx >> 5], 1.0f);
}

// MLP head: anomaly[G] then emb[G,64] into out
__global__ void head_kernel(const float* __restrict__ pooled,
                            const float* __restrict__ A1, const float* __restrict__ ba1,
                            const float* __restrict__ A2, const float* __restrict__ ba2,
                            const float* __restrict__ A3, const float* __restrict__ ba3,
                            const float* __restrict__ Wemb, const float* __restrict__ bemb,
                            float* __restrict__ out) {
  int g = threadIdx.x;
  if (g >= GG) return;
  float p[32];
#pragma unroll
  for (int c = 0; c < 32; ++c) p[c] = pooled[g * 32 + c];
  float z1[32];
  for (int c = 0; c < 32; ++c) {
    float s = ba1[c];
    for (int k = 0; k < 32; ++k) s += p[k] * A1[k * 32 + c];
    z1[c] = s > 0.f ? s : 0.f;
  }
  float z2[16];
  for (int c = 0; c < 16; ++c) {
    float s = ba2[c];
    for (int k = 0; k < 32; ++k) s += z1[k] * A2[k * 16 + c];
    z2[c] = s > 0.f ? s : 0.f;
  }
  float s3 = ba3[0];
  for (int k = 0; k < 16; ++k) s3 += z2[k] * A3[k];
  out[g] = 1.0f / (1.0f + __expf(-s3));
  for (int e = 0; e < 64; ++e) {
    float s = bemb[e];
    for (int k = 0; k < 32; ++k) s += p[k] * Wemb[k * 64 + e];
    out[GG + g * 64 + e] = tanhf(s);
  }
}

// ---------------------------------------------------------------------------
// Host launcher
// ---------------------------------------------------------------------------
static inline long long cdivll(long long a, long long b) { return (a + b - 1) / b; }

extern "C" void kernel_launch(void* const* d_in, const int* in_sizes, int n_in,
                              void* d_out, int out_size, void* d_ws, size_t ws_size,
                              hipStream_t stream) {
  (void)in_sizes; (void)n_in; (void)out_size; (void)ws_size;

  const float* x      = (const float*)d_in[0];
  const int*   eidx   = (const int*)  d_in[1];
  const int*   batch  = (const int*)  d_in[2];
  const float* W1     = (const float*)d_in[3];
  const float* a_src1 = (const float*)d_in[4];
  const float* a_dst1 = (const float*)d_in[5];
  const float* b1     = (const float*)d_in[6];
  const float* W2     = (const float*)d_in[7];
  const float* a_src2 = (const float*)d_in[8];
  const float* a_dst2 = (const float*)d_in[9];
  const float* b2     = (const float*)d_in[10];
  const float* Wg     = (const float*)d_in[11];
  const float* bg     = (const float*)d_in[12];
  const float* A1     = (const float*)d_in[13];
  const float* ba1    = (const float*)d_in[14];
  const float* A2     = (const float*)d_in[15];
  const float* ba2    = (const float*)d_in[16];
  const float* A3     = (const float*)d_in[17];
  const float* ba3    = (const float*)d_in[18];
  const float* Wemb   = (const float*)d_in[19];
  const float* bemb   = (const float*)d_in[20];
  float* out = (float*)d_out;

  const int* srcA = eidx;
  const int* dstA = eidx + EE;

  // ---- workspace layout (floats) ----
  float* ws     = (float*)d_ws;
  float* h1lin  = ws;                            // N*256
  float* agg1   = h1lin + (long long)NN * HC1;   // N*256  (becomes h1)
  float* alS1   = agg1  + (long long)NN * HC1;   // N*4
  float* alD1   = alS1  + (long long)NN * H1;
  float* m1     = alD1  + (long long)NN * H1;
  float* den1   = m1    + (long long)NN * H1;
  float* ex1    = den1  + (long long)NN * H1;    // EP*4
  float* h2lin  = ex1   + (long long)EP * H1;    // N*64
  float* agg2   = h2lin + (long long)NN * HID;   // N*64  (becomes h2)
  float* alS2   = agg2  + (long long)NN * HID;   // N
  float* alD2   = alS2  + NN;
  float* m2     = alD2  + NN;
  float* den2   = m2    + NN;
  float* ex2    = den2  + NN;                    // EP
  float* deg    = ex2   + EP;                    // N
  float* dinv   = deg   + NN;                    // N
  float* h3lin  = dinv  + NN;                    // N*32
  float* agg3   = h3lin + (long long)NN * 32;    // N*32  (becomes h3)
  float* sums   = agg3  + (long long)NN * 32;    // G*32
  float* cnts   = sums  + GG * 32;               // G
  float* pooled = cnts  + GG;                    // G*32

  const int TB = 256;
#define FILL(p, v, n) fill_f32<<<(unsigned)cdivll((n), TB), TB, 0, stream>>>((p), (v), (long long)(n))
  FILL(agg1, 0.f, (long long)NN * HC1);
  FILL(m1,  -INFINITY, (long long)NN * H1);
  FILL(den1, 0.f, (long long)NN * H1);
  FILL(agg2, 0.f, (long long)NN * HID);
  FILL(m2,  -INFINITY, (long long)NN);
  FILL(den2, 0.f, (long long)NN);
  FILL(deg,  0.f, (long long)NN);
  FILL(agg3, 0.f, (long long)NN * 32);
  FILL(sums, 0.f, (long long)GG * 32);
  FILL(cnts, 0.f, (long long)GG);
#undef FILL

  const unsigned aggBlocks = (unsigned)cdivll(EP, TB / 32);   // 1 wave / edge

  // ---- GAT layer 1 ----
  gemm_wmma_f16<<<dim3(NN / 16, HC1 / (16 * WPB)), 32 * WPB,
                  (size_t)(WPB * 16) * F_IN * sizeof(_Float16), stream>>>(
      x, W1, h1lin, NN, F_IN, HC1);
  attn_logits<H1, HID><<<(unsigned)cdivll((long long)NN * H1, TB), TB, 0, stream>>>(
      h1lin, a_src1, a_dst1, alS1, alD1);
  edge_max<H1><<<(unsigned)cdivll((long long)EP * H1, TB), TB, 0, stream>>>(
      srcA, dstA, alS1, alD1, m1);
  edge_expsum<H1><<<(unsigned)cdivll((long long)EP * H1, TB), TB, 0, stream>>>(
      srcA, dstA, alS1, alD1, m1, ex1, den1);
  alpha_div<H1><<<(unsigned)cdivll((long long)EP * H1, TB), TB, 0, stream>>>(
      srcA, dstA, ex1, den1);
  edge_aggregate<H1, HID><<<aggBlocks, TB, 0, stream>>>(srcA, dstA, h1lin, ex1, agg1);
  bias_relu<HC1><<<(unsigned)cdivll((long long)NN * HC1, TB), TB, 0, stream>>>(agg1, b1);

  // ---- GAT layer 2 (heads=1) ----
  gemm_wmma_f16<<<dim3(NN / 16, 1), 32 * WPB,
                  (size_t)(WPB * 16) * HC1 * sizeof(_Float16), stream>>>(
      agg1, W2, h2lin, NN, HC1, HID);
  attn_logits<1, HID><<<(unsigned)cdivll((long long)NN, TB), TB, 0, stream>>>(
      h2lin, a_src2, a_dst2, alS2, alD2);
  edge_max<1><<<(unsigned)cdivll((long long)EP, TB), TB, 0, stream>>>(
      srcA, dstA, alS2, alD2, m2);
  edge_expsum<1><<<(unsigned)cdivll((long long)EP, TB), TB, 0, stream>>>(
      srcA, dstA, alS2, alD2, m2, ex2, den2);
  alpha_div<1><<<(unsigned)cdivll((long long)EP, TB), TB, 0, stream>>>(
      srcA, dstA, ex2, den2);
  edge_aggregate<1, HID><<<aggBlocks, TB, 0, stream>>>(srcA, dstA, h2lin, ex2, agg2);
  bias_relu<HID><<<(unsigned)cdivll((long long)NN * HID, TB), TB, 0, stream>>>(agg2, b2);

  // ---- GCN layer ----
  degree_count<<<(unsigned)cdivll(EP, TB), TB, 0, stream>>>(srcA, dstA, deg);
  deg_rsqrt<<<(unsigned)cdivll(NN, TB), TB, 0, stream>>>(deg, dinv);
  gemm_wmma_f16<<<dim3(NN / 16, 1), 32 * WPB,
                  (size_t)(WPB * 16) * HID * sizeof(_Float16), stream>>>(
      agg2, Wg, h3lin, NN, HID, 32);
  gcn_aggregate<32><<<aggBlocks, TB, 0, stream>>>(srcA, dstA, h3lin, dinv, agg3);
  bias_relu<32><<<(unsigned)cdivll((long long)NN * 32, TB), TB, 0, stream>>>(agg3, bg);

  // ---- global mean pool + heads ----
  pool_sum<32><<<(unsigned)cdivll((long long)NN * 32, TB), TB, 0, stream>>>(
      agg3, batch, sums, cnts);
  pool_div<<<(unsigned)cdivll((long long)GG * 32, TB), TB, 0, stream>>>(sums, cnts, pooled);
  head_kernel<<<1, 64, 0, stream>>>(pooled, A1, ba1, A2, ba2, A3, ba3, Wemb, bemb, out);
}
